// Rotary_LoFTREncoderLayer_61426622267550
// MI455X (gfx1250) — matrix operand order repaired and checked by reading, not compiled, not run on hardware
//
#include <hip/hip_runtime.h>

typedef __attribute__((ext_vector_type(16))) __bf16 v16bf;
typedef __attribute__((ext_vector_type(8)))  __bf16 v8bf;
typedef __attribute__((ext_vector_type(8)))  float  v8f;

#define NB   4
#define LL   8192
#define SS   8192
#define CC   256
#define HH   8
#define DD   32
#define ROWS (NB * LL)          // 32768 token rows (L == S so same for source)

// ---------------------------------------------------------------------------
// helpers
// ---------------------------------------------------------------------------
__device__ inline v16bf combine16(v8bf lo, v8bf hi) {
  v16bf r;
#pragma unroll
  for (int i = 0; i < 8; ++i) { r[i] = lo[i]; r[i + 8] = hi[i]; }
  return r;
}

// Load one 16x32 A fragment and four 32x16 B fragments for k-step k0.
__device__ inline void load_frags(const __bf16* __restrict__ arow,
                                  const __bf16* __restrict__ W, int K,
                                  int n0, int lm, int hi, int k0,
                                  v16bf& a, v16bf b[4]) {
  // A fragment: halves 0-7 -> K = k0 + 8*hi .. +7 ; halves 8-15 -> +16
  a = combine16(*(const v8bf*)(arow + k0 + hi * 8),
                *(const v8bf*)(arow + k0 + 16 + hi * 8));
#pragma unroll
  for (int u = 0; u < 4; ++u) {
    // B fragment: lane column = n0+u*16+lm, halves 0-7 -> K = k0 + 8*hi .. +7 ; halves 8-15 -> +16
    const __bf16* wrow = W + (size_t)(n0 + u * 16 + lm) * K + k0 + hi * 8;
    b[u] = combine16(*(const v8bf*)(wrow), *(const v8bf*)(wrow + 16));
  }
}

__device__ inline void mma4(v8f acc[4], const v16bf& a, const v16bf b[4]) {
#pragma unroll
  for (int u = 0; u < 4; ++u)
    acc[u] = __builtin_amdgcn_wmma_f32_16x16x32_bf16(false, a, false, b[u],
                                                     (short)0, acc[u], false, false);
}

// ---------------------------------------------------------------------------
// elementwise utility kernels (grid-stride)
// ---------------------------------------------------------------------------
__global__ __launch_bounds__(256) void cast_f32_bf16(const float* __restrict__ s,
                                                     __bf16* __restrict__ d, int n) {
  for (int i = blockIdx.x * blockDim.x + threadIdx.x; i < n; i += gridDim.x * blockDim.x)
    d[i] = (__bf16)s[i];
}

__global__ __launch_bounds__(256) void cast_strided(const float* __restrict__ s,
                                                    __bf16* __restrict__ d,
                                                    int rows, int cols, int ld) {
  int n = rows * cols;
  for (int i = blockIdx.x * blockDim.x + threadIdx.x; i < n; i += gridDim.x * blockDim.x) {
    int r = i / cols, c = i - r * cols;
    d[(size_t)r * ld + c] = (__bf16)s[i];
  }
}

__global__ __launch_bounds__(256) void zero_f32(float* __restrict__ d, int n) {
  for (int i = blockIdx.x * blockDim.x + threadIdx.x; i < n; i += gridDim.x * blockDim.x)
    d[i] = 0.0f;
}

// ---------------------------------------------------------------------------
// Generic GEMM: C[M x Nn] = A[M x K] (bf16, lda) @ W[Nn x K]^T (bf16 row-major)
// wave computes a 16(M) x 64(N) strip; block = 8 waves = 128 rows.
// grid = (Nn/64, M/128). relu!=0 -> ReLU + bf16 store to Cb, else f32 to Cf.
// K is a template parameter so the ping-pong double-buffered k-loop fully
// unrolls: loads for step i+1 issue under the WMMAs of step i with no
// register-rotation moves.
// ---------------------------------------------------------------------------
template <int K>
__global__ __launch_bounds__(256) void gemm_bf16_wmma(
    const __bf16* __restrict__ A, int lda,
    const __bf16* __restrict__ W,
    float* __restrict__ Cf, __bf16* __restrict__ Cb, int ldc, int relu) {
  const int wave = threadIdx.x >> 5, lane = threadIdx.x & 31;
  const int lm = lane & 15, hi = lane >> 4;           // column-in-tile, lane-half
  const int m0 = (blockIdx.y * 8 + wave) * 16;
  const int n0 = blockIdx.x * 64;

  v8f acc[4] = {};
  const __bf16* arow = A + (size_t)(m0 + lm) * lda;

  v16bf aA, bA[4], aB, bB[4];
  load_frags(arow, W, K, n0, lm, hi, 0, aA, bA);
#pragma unroll
  for (int k0 = 0; k0 < K - 64; k0 += 64) {
    load_frags(arow, W, K, n0, lm, hi, k0 + 32, aB, bB);
    mma4(acc, aA, bA);
    load_frags(arow, W, K, n0, lm, hi, k0 + 64, aA, bA);
    mma4(acc, aB, bB);
  }
  load_frags(arow, W, K, n0, lm, hi, K - 32, aB, bB);
  mma4(acc, aA, bA);
  mma4(acc, aB, bB);

#pragma unroll
  for (int u = 0; u < 4; ++u)
#pragma unroll
    for (int r = 0; r < 8; ++r) {
      int row = m0 + hi * 8 + r;
      int col = n0 + u * 16 + lm;
      float v = acc[u][r];
      if (relu) Cb[(size_t)row * ldc + col] = (__bf16)fmaxf(v, 0.0f);
      else      Cf[(size_t)row * ldc + col] = v;
    }
}

// ---------------------------------------------------------------------------
// Q post-processing: elu(q)+1 (f32, masked) and rotary(q) (bf16, masked)
// grid = NB * (LL/32), 128 threads (one channel pair each, 32 rows)
// ---------------------------------------------------------------------------
__global__ __launch_bounds__(128) void post_q(
    const float* __restrict__ qraw, const float* __restrict__ pe,
    const float* __restrict__ mask, float* __restrict__ Qf,
    __bf16* __restrict__ Qpos) {
  const int n = blockIdx.x / (LL / 32);
  const int chunk = blockIdx.x % (LL / 32);
  const int c0 = 2 * threadIdx.x, c1 = c0 + 1;
  for (int i = 0; i < 32; ++i) {
    const int l = chunk * 32 + i;
    const size_t base = ((size_t)n * LL + l) * CC;
    const float m = mask[n * LL + l];
    const float q0 = qraw[base + c0], q1 = qraw[base + c1];
    Qf[base + c0] = (q0 > 0.f ? q0 + 1.f : __expf(q0)) * m;
    Qf[base + c1] = (q1 > 0.f ? q1 + 1.f : __expf(q1)) * m;
    const float cs0 = pe[(base + c0) * 2], sn0 = pe[(base + c0) * 2 + 1];
    const float cs1 = pe[(base + c1) * 2], sn1 = pe[(base + c1) * 2 + 1];
    Qpos[base + c0] = (__bf16)((q0 * cs0 - q1 * sn0) * m);
    Qpos[base + c1] = (__bf16)((q1 * cs1 + q0 * sn1) * m);
  }
}

// ---------------------------------------------------------------------------
// K/V post-processing: K_pos, values (rotary, masked, bf16) and Ksum accumulate
// ---------------------------------------------------------------------------
__global__ __launch_bounds__(128) void post_kv(
    const float* __restrict__ kraw, const float* __restrict__ vraw,
    const float* __restrict__ pe, const float* __restrict__ mask,
    __bf16* __restrict__ Kpos, __bf16* __restrict__ Vals,
    float* __restrict__ Ksum) {
  const int n = blockIdx.x / (SS / 32);
  const int chunk = blockIdx.x % (SS / 32);
  const int c0 = 2 * threadIdx.x, c1 = c0 + 1;
  float s0 = 0.f, s1 = 0.f;
  for (int i = 0; i < 32; ++i) {
    const int s = chunk * 32 + i;
    const size_t base = ((size_t)n * SS + s) * CC;
    const float m = mask[n * SS + s];
    const float k0 = kraw[base + c0], k1 = kraw[base + c1];
    s0 += (k0 > 0.f ? k0 + 1.f : __expf(k0)) * m;
    s1 += (k1 > 0.f ? k1 + 1.f : __expf(k1)) * m;
    const float cs0 = pe[(base + c0) * 2], sn0 = pe[(base + c0) * 2 + 1];
    const float cs1 = pe[(base + c1) * 2], sn1 = pe[(base + c1) * 2 + 1];
    Kpos[base + c0] = (__bf16)((k0 * cs0 - k1 * sn0) * m);
    Kpos[base + c1] = (__bf16)((k1 * cs1 + k0 * sn1) * m);
    const float v0 = vraw[base + c0], v1 = vraw[base + c1];
    Vals[base + c0] = (__bf16)((v0 * cs0 - v1 * sn0) * m);
    Vals[base + c1] = (__bf16)((v1 * cs1 + v0 * sn1) * m);
  }
  atomicAdd(&Ksum[n * CC + c0], s0);
  atomicAdd(&Ksum[n * CC + c1], s1);
}

// ---------------------------------------------------------------------------
// KV[n,h,d,v] = sum_s Kpos[n,s,h,d] * Vals[n,s,h,v]  (D x S @ S x D, WMMA)
// grid = (8 splits, NB*HH). 8 waves, each wave 4 k-steps of 32 s.
// ---------------------------------------------------------------------------
__global__ __launch_bounds__(256) void kv_accum(
    const __bf16* __restrict__ Kpos, const __bf16* __restrict__ Vals,
    float* __restrict__ KV) {
  const int wave = threadIdx.x >> 5, lane = threadIdx.x & 31;
  const int lm = lane & 15, hi = lane >> 4;
  const int nh = blockIdx.y, n = nh >> 3, h = nh & 7;
  v8f acc[2][2] = {};
  for (int it = 0; it < 4; ++it) {
    const int s0 = blockIdx.x * 1024 + wave * 128 + it * 32;
    v16bf a[2];
#pragma unroll
    for (int t = 0; t < 2; ++t) {
      const int d = t * 16 + lm;                        // A row (M = d)
#pragma unroll
      for (int j = 0; j < 16; ++j) {
        const int s = s0 + ((j >> 3) << 4) + hi * 8 + (j & 7);
        a[t][j] = Kpos[((size_t)n * SS + s) * CC + h * DD + d];
      }
    }
#pragma unroll
    for (int u = 0; u < 2; ++u) {
      const int v = u * 16 + lm;                        // B column (N = v)
      v16bf b;
#pragma unroll
      for (int j = 0; j < 16; ++j) {
        const int s = s0 + ((j >> 3) << 4) + hi * 8 + (j & 7);
        b[j] = Vals[((size_t)n * SS + s) * CC + h * DD + v];
      }
#pragma unroll
      for (int t = 0; t < 2; ++t)
        acc[t][u] = __builtin_amdgcn_wmma_f32_16x16x32_bf16(false, a[t], false, b,
                                                            (short)0, acc[t][u], false, false);
    }
  }
#pragma unroll
  for (int t = 0; t < 2; ++t)
#pragma unroll
    for (int u = 0; u < 2; ++u)
#pragma unroll
      for (int r = 0; r < 8; ++r) {
        const int d = t * 16 + hi * 8 + r;
        const int v = u * 16 + lm;
        atomicAdd(&KV[(size_t)nh * (DD * DD) + d * DD + v], acc[t][u][r]);
      }
}

// ---------------------------------------------------------------------------
// Z[n,l,h] = 1 / (dot(Q[n,l,h,:], Ksum[n,h,:]) + 1e-6)
// ---------------------------------------------------------------------------
__global__ __launch_bounds__(256) void z_kernel(const float* __restrict__ Qf,
                                                const float* __restrict__ Ksum,
                                                float* __restrict__ Z) {
  const int idx = blockIdx.x * blockDim.x + threadIdx.x;
  if (idx >= NB * LL * HH) return;
  const int h = idx % HH, nl = idx / HH, n = nl / LL;
  const float* q = Qf + (size_t)nl * CC + h * DD;
  const float* ks = Ksum + (n * HH + h) * DD;
  float s = 0.f;
#pragma unroll
  for (int d = 0; d < DD; ++d) s += q[d] * ks[d];
  Z[idx] = 1.0f / (s + 1e-6f);
}

// ---------------------------------------------------------------------------
// queried[n,l,h,v] = (Qpos[n,l,h,:] @ KV[n,h,:,:]) * Z[n,l,h]   (bf16 out)
// grid = (LL/128, NB*HH); wave handles one 16-row l-tile (16x32x32 -> 2 WMMA)
// ---------------------------------------------------------------------------
__global__ __launch_bounds__(256) void queried_kernel(
    const __bf16* __restrict__ Qpos, const __bf16* __restrict__ KVb,
    const float* __restrict__ Z, __bf16* __restrict__ qrd) {
  const int wave = threadIdx.x >> 5, lane = threadIdx.x & 31;
  const int lm = lane & 15, hi = lane >> 4;
  const int nh = blockIdx.y, n = nh >> 3, h = nh & 7;
  const int l0 = blockIdx.x * 128 + wave * 16;

  const __bf16* ap = Qpos + ((size_t)n * LL + (l0 + lm)) * CC + h * DD + hi * 8;
  v16bf a = combine16(*(const v8bf*)ap, *(const v8bf*)(ap + 16));

  v8f acc[2] = {};
#pragma unroll
  for (int u = 0; u < 2; ++u) {
    const int v = u * 16 + lm;
    v16bf b;
#pragma unroll
    for (int j = 0; j < 16; ++j) {
      const int d = ((j >> 3) << 4) + hi * 8 + (j & 7);
      b[j] = KVb[(size_t)nh * (DD * DD) + d * DD + v];
    }
    acc[u] = __builtin_amdgcn_wmma_f32_16x16x32_bf16(false, a, false, b,
                                                     (short)0, acc[u], false, false);
  }
#pragma unroll
  for (int u = 0; u < 2; ++u)
#pragma unroll
    for (int r = 0; r < 8; ++r) {
      const int l = l0 + hi * 8 + r;
      const float z = Z[((size_t)n * LL + l) * HH + h];
      qrd[((size_t)n * LL + l) * CC + h * DD + u * 16 + lm] = (__bf16)(acc[u][r] * z);
    }
}

// ---------------------------------------------------------------------------
// LayerNorm 1: normalize msg row, write bf16 into cat[:, 256:512]
// ---------------------------------------------------------------------------
__global__ __launch_bounds__(256) void ln1_kernel(const float* __restrict__ in,
                                                  const float* __restrict__ g,
                                                  const float* __restrict__ b,
                                                  __bf16* __restrict__ cat) {
  const int row = blockIdx.x, c = threadIdx.x;
  __shared__ float red[256];
  const float v = in[(size_t)row * CC + c];
  red[c] = v; __syncthreads();
  for (int o = 128; o > 0; o >>= 1) { if (c < o) red[c] += red[c + o]; __syncthreads(); }
  const float mean = red[0] * (1.0f / CC); __syncthreads();
  const float d = v - mean;
  red[c] = d * d; __syncthreads();
  for (int o = 128; o > 0; o >>= 1) { if (c < o) red[c] += red[c + o]; __syncthreads(); }
  const float var = red[0] * (1.0f / CC);
  const float y = d * rsqrtf(var + 1e-5f) * g[c] + b[c];
  cat[(size_t)row * (2 * CC) + CC + c] = (__bf16)y;
}

// ---------------------------------------------------------------------------
// LayerNorm 2 + residual: out = x + LN(h2)   (f32 out)
// ---------------------------------------------------------------------------
__global__ __launch_bounds__(256) void ln2_kernel(const float* __restrict__ in,
                                                  const float* __restrict__ g,
                                                  const float* __restrict__ b,
                                                  const float* __restrict__ x,
                                                  float* __restrict__ out) {
  const int row = blockIdx.x, c = threadIdx.x;
  __shared__ float red[256];
  const float v = in[(size_t)row * CC + c];
  red[c] = v; __syncthreads();
  for (int o = 128; o > 0; o >>= 1) { if (c < o) red[c] += red[c + o]; __syncthreads(); }
  const float mean = red[0] * (1.0f / CC); __syncthreads();
  const float d = v - mean;
  red[c] = d * d; __syncthreads();
  for (int o = 128; o > 0; o >>= 1) { if (c < o) red[c] += red[c + o]; __syncthreads(); }
  const float var = red[0] * (1.0f / CC);
  const float y = d * rsqrtf(var + 1e-5f) * g[c] + b[c];
  out[(size_t)row * CC + c] = x[(size_t)row * CC + c] + y;
}

// ---------------------------------------------------------------------------
// host launcher
// ---------------------------------------------------------------------------
extern "C" void kernel_launch(void* const* d_in, const int* in_sizes, int n_in,
                              void* d_out, int out_size, void* d_ws, size_t ws_size,
                              hipStream_t stream) {
  const float* x        = (const float*)d_in[0];
  const float* source   = (const float*)d_in[1];
  const float* x_pe     = (const float*)d_in[2];
  const float* src_pe   = (const float*)d_in[3];
  const float* x_mask   = (const float*)d_in[4];
  const float* src_mask = (const float*)d_in[5];
  const float* Wq = (const float*)d_in[6];
  const float* Wk = (const float*)d_in[7];
  const float* Wv = (const float*)d_in[8];
  const float* Wm = (const float*)d_in[9];
  const float* W1 = (const float*)d_in[10];
  const float* W2 = (const float*)d_in[11];
  const float* g1 = (const float*)d_in[12];
  const float* b1 = (const float*)d_in[13];
  const float* g2 = (const float*)d_in[14];
  const float* b2 = (const float*)d_in[15];
  float* out = (float*)d_out;

  // bump allocator over workspace
  char* p = (char*)d_ws;
  auto alloc = [&](size_t bytes) -> void* {
    void* r = (void*)p; p += (bytes + 255) & ~(size_t)255; return r;
  };
  __bf16* Wq_b = (__bf16*)alloc((size_t)CC * CC * 2);
  __bf16* Wk_b = (__bf16*)alloc((size_t)CC * CC * 2);
  __bf16* Wv_b = (__bf16*)alloc((size_t)CC * CC * 2);
  __bf16* Wm_b = (__bf16*)alloc((size_t)CC * CC * 2);
  __bf16* W1_b = (__bf16*)alloc((size_t)512 * 512 * 2);
  __bf16* W2_b = (__bf16*)alloc((size_t)256 * 512 * 2);
  __bf16* cat  = (__bf16*)alloc((size_t)ROWS * 512 * 2);   // [x | ln1(msg)] bf16
  __bf16* srcb = (__bf16*)alloc((size_t)ROWS * CC * 2);
  float*  qraw = (float*)alloc((size_t)ROWS * CC * 4);     // later: msg_raw
  float*  kraw = (float*)alloc((size_t)ROWS * CC * 4);     // later: h_bf (bf16 ROWSx512)
  float*  vraw = (float*)alloc((size_t)ROWS * CC * 4);     // later: out2
  float*  Qf   = (float*)alloc((size_t)ROWS * CC * 4);
  __bf16* Qpos = (__bf16*)alloc((size_t)ROWS * CC * 2);
  __bf16* Kpos = (__bf16*)alloc((size_t)ROWS * CC * 2);
  __bf16* Vals = (__bf16*)alloc((size_t)ROWS * CC * 2);
  float*  Ksum = (float*)alloc((size_t)NB * CC * 4);       // 1024 f32
  float*  KV   = (float*)alloc((size_t)NB * HH * DD * DD * 4); // 32768 f32
  __bf16* KVb  = (__bf16*)alloc((size_t)NB * HH * DD * DD * 2);
  float*  Zb   = (float*)alloc((size_t)NB * LL * HH * 4);
  __bf16* qrd  = (__bf16*)alloc((size_t)ROWS * CC * 2);

  float*  msg_raw = qraw;            // alias (qraw dead after post_q)
  __bf16* h_bf    = (__bf16*)kraw;   // alias (kraw dead after post_kv)
  float*  out2    = vraw;            // alias (vraw dead after post_kv)

  // --- prep: casts + zeros ---
  cast_f32_bf16<<<256, 256, 0, stream>>>(Wq, Wq_b, CC * CC);
  cast_f32_bf16<<<256, 256, 0, stream>>>(Wk, Wk_b, CC * CC);
  cast_f32_bf16<<<256, 256, 0, stream>>>(Wv, Wv_b, CC * CC);
  cast_f32_bf16<<<256, 256, 0, stream>>>(Wm, Wm_b, CC * CC);
  cast_f32_bf16<<<512, 256, 0, stream>>>(W1, W1_b, 512 * 512);
  cast_f32_bf16<<<512, 256, 0, stream>>>(W2, W2_b, 256 * 512);
  cast_strided<<<2048, 256, 0, stream>>>(x, cat, ROWS, CC, 512);     // cat[:, :256] = x
  cast_f32_bf16<<<2048, 256, 0, stream>>>(source, srcb, ROWS * CC);
  zero_f32<<<144, 256, 0, stream>>>(Ksum, NB * CC + NB * HH * DD * DD); // Ksum+KV contiguous

  // --- projections: q = x@Wq^T, k = src@Wk^T, v = src@Wv^T ---
  gemm_bf16_wmma<256><<<dim3(CC / 64, ROWS / 128), 256, 0, stream>>>(cat,  512, Wq_b, qraw, nullptr, CC, 0);
  gemm_bf16_wmma<256><<<dim3(CC / 64, ROWS / 128), 256, 0, stream>>>(srcb, CC,  Wk_b, kraw, nullptr, CC, 0);
  gemm_bf16_wmma<256><<<dim3(CC / 64, ROWS / 128), 256, 0, stream>>>(srcb, CC,  Wv_b, vraw, nullptr, CC, 0);

  // --- elementwise post-processing ---
  post_q <<<NB * (LL / 32), 128, 0, stream>>>(qraw, x_pe, x_mask, Qf, Qpos);
  post_kv<<<NB * (SS / 32), 128, 0, stream>>>(kraw, vraw, src_pe, src_mask, Kpos, Vals, Ksum);

  // --- KV = K_pos^T @ values (per n,h), then bf16 cast ---
  kv_accum<<<dim3(8, NB * HH), 256, 0, stream>>>(Kpos, Vals, KV);
  cast_f32_bf16<<<128, 256, 0, stream>>>(KV, KVb, NB * HH * DD * DD);

  // --- Z then queried ---
  z_kernel<<<(NB * LL * HH) / 256, 256, 0, stream>>>(Qf, Ksum, Zb);
  queried_kernel<<<dim3(LL / 128, NB * HH), 256, 0, stream>>>(Qpos, KVb, Zb, qrd);

  // --- message = queried @ Wm^T ; LN1 -> cat[:, 256:] ---
  gemm_bf16_wmma<256><<<dim3(CC / 64, ROWS / 128), 256, 0, stream>>>(qrd, CC, Wm_b, msg_raw, nullptr, CC, 0);
  ln1_kernel<<<ROWS, 256, 0, stream>>>(msg_raw, g1, b1, cat);

  // --- MLP: h = relu(cat @ W1^T) (bf16) ; out2 = h @ W2^T (f32) ---
  gemm_bf16_wmma<512><<<dim3(512 / 64, ROWS / 128), 256, 0, stream>>>(cat, 512, W1_b, nullptr, h_bf, 512, 1);
  gemm_bf16_wmma<512><<<dim3(CC / 64, ROWS / 128), 256, 0, stream>>>(h_bf, 512, W2_b, out2, nullptr, CC, 0);

  // --- LN2 + residual ---
  ln2_kernel<<<ROWS, 256, 0, stream>>>(out2, g2, b2, x, out);
}
